// GatedMultiHeadAttention_59854664237085
// MI455X (gfx1250) — matrix operand order repaired and checked by reading, not compiled, not run on hardware
//
#include <hip/hip_runtime.h>
#include <hip/hip_bf16.h>

// GatedMultiHeadAttention for MI455X (gfx1250), wave32 + WMMA bf16 + TDM.
// B=4, N=2048, D=1024, H=16, dk=64.

#define BATCH   4
#define SEQ     2048
#define DMODEL  1024
#define HEADS   16
#define DK      64
#define MTOT    (BATCH * SEQ)          // 8192
#define BSTRIDE 20                     // padded [n][kpair] stride (u32 units)

typedef __attribute__((ext_vector_type(16))) __bf16 bf16x16;
typedef __attribute__((ext_vector_type(8)))  float  floatx8;
typedef __attribute__((ext_vector_type(4)))  unsigned int uint32x4;
typedef __attribute__((ext_vector_type(8)))  int    int32x8;
typedef __attribute__((ext_vector_type(4)))  int    int32x4;

union Frag {
    bf16x16      h;
    unsigned int u[8];
};

__device__ __forceinline__ floatx8 wmma_bf16(const Frag& a, const Frag& b, floatx8 c) {
    return __builtin_amdgcn_wmma_f32_16x16x32_bf16(
        /*neg_a=*/false, a.h, /*neg_b=*/false, b.h,
        /*c_mod=*/(short)0, c, /*reuse_a=*/false, /*reuse_b=*/false);
}

__device__ __forceinline__ unsigned int pack_bf16x2(float a, float b) {
    unsigned short lo = __builtin_bit_cast(unsigned short, (__bf16)a);
    unsigned short hi = __builtin_bit_cast(unsigned short, (__bf16)b);
    return ((unsigned int)hi << 16) | (unsigned int)lo;
}

#define NEG_INF (-__builtin_inff())

// ---------------------------------------------------------------------------
// Tensor Data Mover: 2D bf16 tile load, global -> LDS (contiguous row-major).
// Builds the D# per CDNA5 ISA and issues
// TENSOR_LOAD_TO_LDS.  Caller must be wave-uniform; pair with
// s_wait_tensorcnt(0) on the issuing wave + a block barrier.
// ---------------------------------------------------------------------------
__device__ __forceinline__ void tdm_load_2d_bf16(unsigned lds_off, const void* gptr,
                                                 unsigned tile_d0, unsigned tile_d1,
                                                 unsigned stride0)
{
    const unsigned long long ga = (unsigned long long)(uintptr_t)gptr;
    uint32x4 g0;
    g0[0] = 1u;                                   // count=1, user descriptor
    g0[1] = lds_off;                              // LDS byte address
    g0[2] = (unsigned)(ga & 0xffffffffu);         // global_addr[31:0]
    g0[3] = (unsigned)((ga >> 32) & 0x01ffffffu)  // global_addr[56:32]
          | (2u << 30);                           // type = 2 ("image")
    int32x8 g1;
    g1[0] = (int)(1u << 16);                      // data_size = 1 -> 2 bytes
    g1[1] = (int)((tile_d0 & 0xffffu) << 16);     // tensor_dim0[15:0]
    g1[2] = (int)(((tile_d0 >> 16) & 0xffffu) |   // tensor_dim0[31:16]
                  ((tile_d1 & 0xffffu) << 16));   // tensor_dim1[15:0]
    g1[3] = (int)(((tile_d1 >> 16) & 0xffffu) |   // tensor_dim1[31:16]
                  ((tile_d0 & 0xffffu) << 16));   // tile_dim0
    g1[4] = (int)(tile_d1 & 0xffffu);             // tile_dim1 (tile_dim2 = 0)
    g1[5] = (int)stride0;                         // tensor_dim0_stride[31:0]
    g1[6] = 0;
    g1[7] = 0;
    const int32x4 z4 = {0, 0, 0, 0};
#if __clang_major__ >= 23
    const int32x8 z8 = {0, 0, 0, 0, 0, 0, 0, 0};
    __builtin_amdgcn_tensor_load_to_lds(g0, g1, z4, z4, z8, 0);
#else
    __builtin_amdgcn_tensor_load_to_lds(g0, g1, z4, z4, 0);
#endif
}

__device__ __forceinline__ unsigned lds_offset(const void* p) {
    // Generic-pointer low 32 bits == LDS byte offset (aperture check truncation).
    return (unsigned)(uintptr_t)p;
}

// ---------------------------------------------------------------------------
// Kernel 1: fused Q/K/V/Gate projections.  Y = x @ W + b  (gate: sigmoid).
// Output: bf16 (B, H, N, dk).  grid = (MTOT/128, DMODEL/64, 4), block = 128.
// Wave w computes a 32x64 sub-tile (2x4 fragments).  A staged as k-pair u32;
// B staged transposed [n][kpair] with padded stride so fragments are b128.
// ---------------------------------------------------------------------------
__global__ __launch_bounds__(128)
void proj_kernel(const float* __restrict__ x,
                 const float* __restrict__ Wq, const float* __restrict__ bq,
                 const float* __restrict__ Wk, const float* __restrict__ bk,
                 const float* __restrict__ Wv, const float* __restrict__ bv,
                 const float* __restrict__ Wg, const float* __restrict__ bg,
                 __bf16* __restrict__ Qb, __bf16* __restrict__ Kb,
                 __bf16* __restrict__ Vb, __bf16* __restrict__ Gb)
{
    const int mode = blockIdx.z;
    const float* W;  const float* bias;  __bf16* out;
    switch (mode) {
        case 0:  W = Wq; bias = bq; out = Qb; break;
        case 1:  W = Wk; bias = bk; out = Kb; break;
        case 2:  W = Wv; bias = bv; out = Vb; break;
        default: W = Wg; bias = bg; out = Gb; break;
    }

    const int tid  = threadIdx.x;
    const int lane = tid & 31;
    const int wv   = tid >> 5;
    const int half = lane >> 4;
    const int l15  = lane & 15;
    const int row0 = blockIdx.x * 128;
    const int col0 = blockIdx.y * 64;

    __shared__ unsigned int As[128 * 16];          // [m][kpair]       8 KB
    __shared__ unsigned int Bs[64 * BSTRIDE];      // [n][kpair] pad   5 KB

    floatx8 acc[2][4] = {};

    for (int kk = 0; kk < DMODEL; kk += 32) {
        // ---- stage A tile (128x32 f32 -> bf16 pairs) ----
        {
            const float* arow = x + (size_t)(row0 + tid) * DMODEL + kk;
            #pragma unroll
            for (int i = 0; i < 8; ++i) {
                float4 f = ((const float4*)arow)[i];
                As[tid * 16 + 2 * i + 0] = pack_bf16x2(f.x, f.y);
                As[tid * 16 + 2 * i + 1] = pack_bf16x2(f.z, f.w);
            }
        }
        // ---- stage B tile transposed (32x64 f32 -> [n][kpair] bf16) ----
        {
            const int p  = tid >> 3;       // 0..15  (k pair index)
            const int e  = tid & 7;        // 0..7
            const int n0 = e * 8;
            const float* r0 = W + (size_t)(kk + 2 * p) * DMODEL + col0 + n0;
            const float* r1 = r0 + DMODEL;
            #pragma unroll
            for (int i = 0; i < 2; ++i) {
                float4 a = ((const float4*)r0)[i];
                float4 c = ((const float4*)r1)[i];
                Bs[(n0 + 4 * i + 0) * BSTRIDE + p] = pack_bf16x2(a.x, c.x);
                Bs[(n0 + 4 * i + 1) * BSTRIDE + p] = pack_bf16x2(a.y, c.y);
                Bs[(n0 + 4 * i + 2) * BSTRIDE + p] = pack_bf16x2(a.z, c.z);
                Bs[(n0 + 4 * i + 3) * BSTRIDE + p] = pack_bf16x2(a.w, c.w);
            }
        }
        __syncthreads();

        // ---- fragments + WMMA ----
        Frag a[2], bfr[4];
        #pragma unroll
        for (int i = 0; i < 2; ++i) {
            const int M = wv * 32 + i * 16 + l15;
            #pragma unroll
            for (int r = 0; r < 8; ++r) {
                const int p = half * 8 + r;
                a[i].u[r] = As[M * 16 + p];
            }
        }
        #pragma unroll
        for (int j = 0; j < 4; ++j) {
            const int n = j * 16 + l15;
            #pragma unroll
            for (int r = 0; r < 8; ++r)
                bfr[j].u[r] = Bs[n * BSTRIDE + half * 8 + r];   // 2x ds_load_b128
        }
        #pragma unroll
        for (int i = 0; i < 2; ++i)
            #pragma unroll
            for (int j = 0; j < 4; ++j)
                acc[i][j] = wmma_bf16(a[i], bfr[j], acc[i][j]);
        __syncthreads();
    }

    // ---- epilogue: bias (+sigmoid for gate), write (B,H,N,dk) bf16 ----
    #pragma unroll
    for (int i = 0; i < 2; ++i) {
        #pragma unroll
        for (int j = 0; j < 4; ++j) {
            const int colg = col0 + j * 16 + l15;
            const float bb = bias[colg];
            #pragma unroll
            for (int e = 0; e < 8; ++e) {
                const int rowg = row0 + wv * 32 + i * 16 + e + 8 * half;
                float v = acc[i][j][e] + bb;
                if (mode == 3)
                    v = __builtin_amdgcn_rcpf(1.0f + __expf(-v));
                const int bidx = rowg >> 11;     // / SEQ
                const int nidx = rowg & 2047;
                const int hidx = colg >> 6;      // / DK
                const int didx = colg & 63;
                out[(((size_t)bidx * HEADS + hidx) * SEQ + nidx) * DK + didx] = (__bf16)v;
            }
        }
    }
}

// ---------------------------------------------------------------------------
// Kernel 2: flash-style attention + gate.  One 64-query tile per block,
// wave w owns queries [w*16, w*16+16).  Q and K tiles arrive via the Tensor
// Data Mover (wave 0 issues, s_wait_tensorcnt + barrier publishes); V is
// transposed manually into LDS.  grid = (B*H, SEQ/64), block = 128.
// ---------------------------------------------------------------------------
__global__ __launch_bounds__(128)
void attn_kernel(const __bf16* __restrict__ Qb, const __bf16* __restrict__ Kb,
                 const __bf16* __restrict__ Vb, const __bf16* __restrict__ Gb,
                 const unsigned char* __restrict__ maskp,
                 __bf16* __restrict__ gated)
{
    const int bh = blockIdx.x;
    const int b  = bh >> 4;
    const int h  = bh & 15;
    const int q0 = blockIdx.y * 64;

    const int tid  = threadIdx.x;
    const int lane = tid & 31;
    const int wv   = tid >> 5;
    const int half = lane >> 4;
    const int l15  = lane & 15;

    const size_t headOff = (size_t)bh * SEQ * DK;
    const __bf16* Qh = Qb + headOff;
    const __bf16* Kh = Kb + headOff;
    const __bf16* Vh = Vb + headOff;
    const __bf16* Gh = Gb + headOff;

    __shared__ __attribute__((aligned(128))) unsigned int Qs[64 * 32]; // [q][dkpair]
    __shared__ __attribute__((aligned(128))) unsigned int Ks[64 * 32]; // [key][dkpair]
    __shared__ unsigned int Vt[64 * 32];      // [dim][key pair] (transposed)
    __shared__ unsigned int Ps[4 * 16 * 32];  // per-wave P tile
    __shared__ float maskAdd[64];

    // ---- stage Q tile once via TDM ----
    if (wv == 0) {
        tdm_load_2d_bf16(lds_offset(Qs), Qh + (size_t)q0 * DK, DK, 64, DK);
        __builtin_amdgcn_s_wait_tensorcnt(0);
    }
    __syncthreads();

    // Q fragments (A-layout, 2 k-steps of 32 over dk=64), kept in registers.
    Frag aq[2];
    #pragma unroll
    for (int s = 0; s < 2; ++s) {
        const int M = wv * 16 + l15;
        #pragma unroll
        for (int r = 0; r < 8; ++r) {
            const int k = s * 32 + half * 16 + r * 2;
            aq[s].u[r] = Qs[M * 32 + (k >> 1)];
        }
    }

    floatx8 o[4] = {};
    float m_i[8], l_i[8];
    #pragma unroll
    for (int e = 0; e < 8; ++e) { m_i[e] = NEG_INF; l_i[e] = 0.0f; }

    for (int kt = 0; kt < SEQ; kt += 64) {
        __syncthreads();   // retire previous iteration's LDS readers
        // ---- K tile via TDM (async while V is staged manually) ----
        if (wv == 0)
            tdm_load_2d_bf16(lds_offset(Ks), Kh + (size_t)kt * DK, DK, 64, DK);
        // ---- stage V transposed + mask ----
        {
            const int key = tid >> 1;
            const int hr  = tid & 1;
            union { uint4 v[4]; unsigned short s16[32]; } vb;
            const uint4* srcv = (const uint4*)(Vh + (size_t)(kt + key) * DK + hr * 32);
            vb.v[0] = srcv[0]; vb.v[1] = srcv[1]; vb.v[2] = srcv[2]; vb.v[3] = srcv[3];
            unsigned short* VtU = (unsigned short*)Vt;
            #pragma unroll
            for (int d = 0; d < 32; ++d)
                VtU[(hr * 32 + d) * 64 + key] = vb.s16[d];

            if (tid < 64)
                maskAdd[tid] = maskp[(size_t)b * SEQ + kt + tid] ? NEG_INF : 0.0f;
        }
        if (wv == 0)
            __builtin_amdgcn_s_wait_tensorcnt(0);
        __syncthreads();

        // ---- S = Q @ K^T (2 k-steps x 4 key-fragments) ----
        floatx8 sacc[4] = {};
        #pragma unroll
        for (int j = 0; j < 4; ++j) {
            const int key = j * 16 + l15;
            #pragma unroll
            for (int s = 0; s < 2; ++s) {
                Frag bkf;
                #pragma unroll
                for (int r = 0; r < 8; ++r)
                    bkf.u[r] = Ks[key * 32 + (s * 16 + half * 8 + r)];
                sacc[j] = wmma_bf16(aq[s], bkf, sacc[j]);
            }
        }

        // ---- online softmax (rows = e + 8*half, cols across 16-lane half) ----
        float svv[4][8];
        #pragma unroll
        for (int j = 0; j < 4; ++j)
            #pragma unroll
            for (int e = 0; e < 8; ++e)
                svv[j][e] = sacc[j][e] * 0.125f + maskAdd[j * 16 + l15];

        #pragma unroll
        for (int e = 0; e < 8; ++e) {
            float mx = fmaxf(fmaxf(svv[0][e], svv[1][e]), fmaxf(svv[2][e], svv[3][e]));
            mx = fmaxf(mx, __shfl_xor(mx, 1));
            mx = fmaxf(mx, __shfl_xor(mx, 2));
            mx = fmaxf(mx, __shfl_xor(mx, 4));
            mx = fmaxf(mx, __shfl_xor(mx, 8));
            const float mnew = fmaxf(m_i[e], mx);
            const float sc   = __expf(m_i[e] - mnew);
            float rs = 0.0f;
            #pragma unroll
            for (int j = 0; j < 4; ++j) {
                const float p = __expf(svv[j][e] - mnew);
                svv[j][e] = p;
                rs += p;
            }
            rs += __shfl_xor(rs, 1);
            rs += __shfl_xor(rs, 2);
            rs += __shfl_xor(rs, 4);
            rs += __shfl_xor(rs, 8);
            l_i[e] = l_i[e] * sc + rs;
            m_i[e] = mnew;
            #pragma unroll
            for (int df = 0; df < 4; ++df) o[df][e] *= sc;
        }

        // ---- C-layout -> A-layout via per-wave LDS patch ----
        {
            unsigned short* PsU = (unsigned short*)&Ps[wv * 16 * 32];
            #pragma unroll
            for (int j = 0; j < 4; ++j)
                #pragma unroll
                for (int e = 0; e < 8; ++e)
                    PsU[(e + 8 * half) * 64 + j * 16 + l15] =
                        __builtin_bit_cast(unsigned short, (__bf16)svv[j][e]);
        }
        __syncthreads();

        // ---- O += P @ V ----
        const unsigned int* PsW = &Ps[wv * 16 * 32];
        #pragma unroll
        for (int s = 0; s < 2; ++s) {
            Frag ap;
            #pragma unroll
            for (int r = 0; r < 8; ++r) {
                const int k = s * 32 + half * 16 + r * 2;
                ap.u[r] = PsW[l15 * 32 + (k >> 1)];
            }
            #pragma unroll
            for (int df = 0; df < 4; ++df) {
                Frag bvf;
                const int dim = df * 16 + l15;
                #pragma unroll
                for (int r = 0; r < 8; ++r)
                    bvf.u[r] = Vt[dim * 32 + (s * 16 + half * 8 + r)];
                o[df] = wmma_bf16(ap, bvf, o[df]);
            }
        }
    }

    // ---- epilogue: normalize, gate, write (B, N, D) bf16 ----
    #pragma unroll
    for (int e = 0; e < 8; ++e) {
        const float linv = __builtin_amdgcn_rcpf(l_i[e]);
        const int qg = q0 + wv * 16 + e + 8 * half;
        #pragma unroll
        for (int df = 0; df < 4; ++df) {
            const int dim = df * 16 + l15;
            const float g = (float)Gh[(size_t)qg * DK + dim];
            const float v = o[df][e] * linv * g;
            gated[((size_t)(b * SEQ + qg)) * DMODEL + h * DK + dim] = (__bf16)v;
        }
    }
}

// ---------------------------------------------------------------------------
// Kernel 3: out = gated(bf16) @ Wo + bo, f32 output.  A tile (already bf16,
// contiguous k-pairs) arrives via TDM; B transposed in LDS as in kernel 1.
// grid = (MTOT/128, DMODEL/64), block = 128.
// ---------------------------------------------------------------------------
__global__ __launch_bounds__(128)
void out_proj_kernel(const __bf16* __restrict__ gated,
                     const float* __restrict__ Wo, const float* __restrict__ bo,
                     float* __restrict__ outp)
{
    const int tid  = threadIdx.x;
    const int lane = tid & 31;
    const int wv   = tid >> 5;
    const int half = lane >> 4;
    const int l15  = lane & 15;
    const int row0 = blockIdx.x * 128;
    const int col0 = blockIdx.y * 64;

    __shared__ __attribute__((aligned(128))) unsigned int As[128 * 16]; // [m][kpair]
    __shared__ unsigned int Bs[64 * BSTRIDE];

    floatx8 acc[2][4] = {};

    for (int kk = 0; kk < DMODEL; kk += 32) {
        // ---- A tile (128 rows x 32 bf16) via TDM: layout matches [m][kpair] ----
        if (wv == 0)
            tdm_load_2d_bf16(lds_offset(As),
                             gated + (size_t)row0 * DMODEL + kk,
                             /*tile_d0=*/32, /*tile_d1=*/128, /*stride0=*/DMODEL);
        // ---- stage B transposed ----
        {
            const int p  = tid >> 3;
            const int e  = tid & 7;
            const int n0 = e * 8;
            const float* r0 = Wo + (size_t)(kk + 2 * p) * DMODEL + col0 + n0;
            const float* r1 = r0 + DMODEL;
            #pragma unroll
            for (int i = 0; i < 2; ++i) {
                float4 a = ((const float4*)r0)[i];
                float4 c = ((const float4*)r1)[i];
                Bs[(n0 + 4 * i + 0) * BSTRIDE + p] = pack_bf16x2(a.x, c.x);
                Bs[(n0 + 4 * i + 1) * BSTRIDE + p] = pack_bf16x2(a.y, c.y);
                Bs[(n0 + 4 * i + 2) * BSTRIDE + p] = pack_bf16x2(a.z, c.z);
                Bs[(n0 + 4 * i + 3) * BSTRIDE + p] = pack_bf16x2(a.w, c.w);
            }
        }
        if (wv == 0)
            __builtin_amdgcn_s_wait_tensorcnt(0);
        __syncthreads();

        Frag a[2], bfr[4];
        #pragma unroll
        for (int i = 0; i < 2; ++i) {
            const int M = wv * 32 + i * 16 + l15;
            #pragma unroll
            for (int r = 0; r < 8; ++r) {
                const int p = half * 8 + r;
                a[i].u[r] = As[M * 16 + p];
            }
        }
        #pragma unroll
        for (int j = 0; j < 4; ++j) {
            const int n = j * 16 + l15;
            #pragma unroll
            for (int r = 0; r < 8; ++r)
                bfr[j].u[r] = Bs[n * BSTRIDE + half * 8 + r];
        }
        #pragma unroll
        for (int i = 0; i < 2; ++i)
            #pragma unroll
            for (int j = 0; j < 4; ++j)
                acc[i][j] = wmma_bf16(a[i], bfr[j], acc[i][j]);
        __syncthreads();
    }

    #pragma unroll
    for (int i = 0; i < 2; ++i) {
        #pragma unroll
        for (int j = 0; j < 4; ++j) {
            const int colg = col0 + j * 16 + l15;
            const float bb = bo[colg];
            #pragma unroll
            for (int e = 0; e < 8; ++e) {
                const int rowg = row0 + wv * 32 + i * 16 + e + 8 * half;
                outp[(size_t)rowg * DMODEL + colg] = acc[i][j][e] + bb;
            }
        }
    }
}

// ---------------------------------------------------------------------------
extern "C" void kernel_launch(void* const* d_in, const int* in_sizes, int n_in,
                              void* d_out, int out_size, void* d_ws, size_t ws_size,
                              hipStream_t stream)
{
    (void)in_sizes; (void)n_in; (void)out_size; (void)ws_size;

    const float*         x    = (const float*)d_in[0];
    const unsigned char* mask = (const unsigned char*)d_in[1];
    const float* Wq = (const float*)d_in[2];
    const float* bq = (const float*)d_in[3];
    const float* Wk = (const float*)d_in[4];
    const float* bk = (const float*)d_in[5];
    const float* Wv = (const float*)d_in[6];
    const float* bv = (const float*)d_in[7];
    const float* Wo = (const float*)d_in[8];
    const float* bo = (const float*)d_in[9];
    const float* Wg = (const float*)d_in[10];
    const float* bg = (const float*)d_in[11];
    float* outp = (float*)d_out;

    // Workspace: Q,K,V,Gate in (B,H,N,dk) bf16 + gated in (B,N,D) bf16 = 80 MB.
    const size_t SZ = (size_t)MTOT * DMODEL;
    __bf16* Qb    = (__bf16*)d_ws;
    __bf16* Kb    = Qb + SZ;
    __bf16* Vb    = Kb + SZ;
    __bf16* Gb    = Vb + SZ;
    __bf16* gated = Gb + SZ;

    dim3 blk(128);
    proj_kernel<<<dim3(MTOT / 128, DMODEL / 64, 4), blk, 0, stream>>>(
        x, Wq, bq, Wk, bk, Wv, bv, Wg, bg, Qb, Kb, Vb, Gb);
    attn_kernel<<<dim3(BATCH * HEADS, SEQ / 64), blk, 0, stream>>>(
        Qb, Kb, Vb, Gb, mask, gated);
    out_proj_kernel<<<dim3(MTOT / 128, DMODEL / 64), blk, 0, stream>>>(
        gated, Wo, bo, outp);
}
